// VectorizedObjectSelector_61770219651143
// MI455X (gfx1250) — matrix operand order repaired and checked by reading, not compiled, not run on hardware
//
#include <hip/hip_runtime.h>
#include <hip/hip_bf16.h>

// Shapes from the reference
#define B_N   4096
#define K_N   64
#define D_N   512

typedef _Float16 v16h __attribute__((ext_vector_type(16)));
typedef _Float16 v4h  __attribute__((ext_vector_type(4)));
typedef float    v8f  __attribute__((ext_vector_type(8)));

// Pack 8 floats (two float4) into 8 consecutive f16 slots of a v16h.
__device__ __forceinline__ void pack8(v16h& d, int base, float4 p, float4 q) {
    d[base + 0] = (_Float16)p.x; d[base + 1] = (_Float16)p.y;
    d[base + 2] = (_Float16)p.z; d[base + 3] = (_Float16)p.w;
    d[base + 4] = (_Float16)q.x; d[base + 5] = (_Float16)q.y;
    d[base + 6] = (_Float16)q.z; d[base + 7] = (_Float16)q.w;
}

// One block per batch element b. 128 threads = 4 wave32; wave w computes
// scores[b, 16w .. 16w+15] via 16 chained v_wmma_f32_16x16x32_f16 ops over
// the D=512 reduction (16 chunks of 32). The query vector v[b,:] is staged
// once per block into LDS as f16 and shared by all 4 waves.
__global__ __launch_bounds__(128) void VectorizedObjectSelector_wmma_kernel(
    const float* __restrict__ vectors,     // (B, D) f32
    const int*   __restrict__ impls,       // (B, K) indices
    const float* __restrict__ embed,       // (VOCAB, D) f32
    float* __restrict__ out_idx,           // (B, K) -> indices as float
    float* __restrict__ out_scores)        // (B, K) -> scores
{
    const int b    = blockIdx.x;
    const int lane = threadIdx.x & 31;
    const int wave = threadIdx.x >> 5;       // 0..3
    const int kbase = wave * 16;
    const int m    = lane & 15;              // matrix row handled by this lane
    const bool hi  = (lane >= 16);           // high half-wave holds the other K-halves

    const float* vrow = vectors + (size_t)b * D_N;   // query vector

    // ---- Stage v[b,:] into LDS as f16 (1 KB), converted once per block ----
    __shared__ __align__(32) _Float16 vsh[D_N];
    {
        const int i = (int)threadIdx.x * 4;          // 128 threads * 4 = 512
        float4 t = *(const float4*)(vrow + i);
        v4h h;
        h[0] = (_Float16)t.x; h[1] = (_Float16)t.y;
        h[2] = (_Float16)t.z; h[3] = (_Float16)t.w;
        *(v4h*)(vsh + i) = h;
    }

    // Gather index for row m of this wave's 16x512 tile (overlaps LDS store).
    const int idx = impls[(size_t)b * K_N + kbase + m];
    if (!hi) out_idx[(size_t)b * K_N + kbase + m] = (float)idx;

    const float* erow = embed + (size_t)idx * D_N;   // gathered embedding row

    __syncthreads();

    // Per-lane K offsets for the 16-bit WMMA A operand (16x32):
    //   lanes 0-15 : K = {0..15}
    //   lanes 16-31: K = {16..31}
    const int ka0 = hi ? 16 : 0;
    const int ka1 = hi ? 24 : 8;
    // B layout (32x16): lanes 0-15 hold K=0..15, lanes 16-31 hold K=16..31.
    const int kb  = hi ? 16 : 0;

    v8f acc = {};

#pragma unroll
    for (int c = 0; c < D_N / 32; ++c) {
        const float* ea = erow + c * 32;

        // A-operand: 16 f16 from this lane's gathered row (32B + 32B contiguous).
        float4 a0 = *(const float4*)(ea + ka0);
        float4 a1 = *(const float4*)(ea + ka0 + 4);
        float4 a2 = *(const float4*)(ea + ka1);
        float4 a3 = *(const float4*)(ea + ka1 + 4);

        // B-operand: 16 contiguous f16 of v from LDS (2 x ds_load_b128);
        // replicates v into all 16 output columns.
        const v16h Bm = *(const v16h*)(vsh + c * 32 + kb);

        v16h Am;
        pack8(Am, 0, a0, a1);
        pack8(Am, 8, a2, a3);

        // D = A(16x32) * B(32x16) + C ; accumulates the dot over D in f32.
        acc = __builtin_amdgcn_wmma_f32_16x16x32_f16(
            /*neg_a=*/false, Am, /*neg_b=*/false, Bm,
            /*c_mod=*/(short)0, acc, /*reuse_a=*/false, /*reuse_b=*/false);
    }

    // f32 C/D 16x16 layout: lane n (0-15) VGPR r = D[M=r, N=n];
    // lane n+16 VGPR r = D[M=8+r, N=n]. Every column is identical, so
    // column 0 (lanes 0 and 16) carries all 16 scores.
    if (m == 0) {
        float* o = out_scores + (size_t)b * K_N + kbase + (hi ? 8 : 0);
        float4 lo = make_float4(acc[0], acc[1], acc[2], acc[3]);
        float4 hi4 = make_float4(acc[4], acc[5], acc[6], acc[7]);
        *(float4*)(o + 0) = lo;
        *(float4*)(o + 4) = hi4;
    }
}

extern "C" void kernel_launch(void* const* d_in, const int* in_sizes, int n_in,
                              void* d_out, int out_size, void* d_ws, size_t ws_size,
                              hipStream_t stream) {
    const float* vectors = (const float*)d_in[0];   // (B, D) f32
    const int*   impls   = (const int*)d_in[1];     // (B, K) int
    const float* embed   = (const float*)d_in[2];   // (VOCAB, D) f32

    float* out_idx    = (float*)d_out;                       // first B*K
    float* out_scores = out_idx + (size_t)B_N * K_N;         // second B*K

    VectorizedObjectSelector_wmma_kernel<<<B_N, 128, 0, stream>>>(
        vectors, impls, embed, out_idx, out_scores);
}
